// HGRNBitMLP_38886633898328
// MI455X (gfx1250) — 8-bit matrix operand order repaired and checked by reading, compiled, not run on hardware
//
#include <hip/hip_runtime.h>
#include <hip/hip_bf16.h>


typedef int v8i __attribute__((ext_vector_type(8)));

#define TOKENS 4096
#define HDIM   2048
#define IDIM   8192

__device__ __forceinline__ v8i zero_v8i() {
    v8i z;
#pragma unroll
    for (int i = 0; i < 8; ++i) z[i] = 0;
    return z;
}

__device__ __forceinline__ void async_b128(unsigned lds, int voff, const signed char* base) {
    asm volatile("global_load_async_to_lds_b128 %0, %1, %2"
                 :: "v"(lds), "v"(voff), "s"(base) : "memory");
}
__device__ __forceinline__ void async_b128_off16(unsigned lds, int voff, const signed char* base) {
    asm volatile("global_load_async_to_lds_b128 %0, %1, %2 offset:16"
                 :: "v"(lds), "v"(voff), "s"(base) : "memory");
}


__global__ void init_red_kernel(float* red) {
    if (threadIdx.x < 8) red[threadIdx.x] = 0.0f;
}

__global__ void abs_sum_kernel(const float* __restrict__ w, long long n,
                               float* __restrict__ acc) {
    __shared__ float red[256];
    long long i = (long long)blockIdx.x * blockDim.x + threadIdx.x;
    const long long stride = (long long)gridDim.x * blockDim.x;
    float s = 0.0f;
    for (; i < n; i += stride) s += fabsf(w[i]);
    red[threadIdx.x] = s;
    __syncthreads();
    for (int o = 128; o > 0; o >>= 1) {
        if ((int)threadIdx.x < o) red[threadIdx.x] += red[threadIdx.x + o];
        __syncthreads();
    }
    if (threadIdx.x == 0) atomicAdd(acc, red[0]);
}

__global__ void finalize_scales_kernel(float* red, long long n_gate, long long n_down) {
    if (threadIdx.x == 0) {
        float inv_g = fmaxf(red[0] / (float)n_gate, 1e-5f);
        float inv_d = fmaxf(red[1] / (float)n_down, 1e-5f);
        red[2] = 1.0f / inv_g;
        red[3] = inv_g;
        red[4] = 1.0f / inv_d;
        red[5] = inv_d;
    }
}

__global__ void quant_w_kernel(const float* __restrict__ w, signed char* __restrict__ wq,
                               long long n4, const float* __restrict__ s_ptr) {
    const float s = s_ptr[0];
    long long i = (long long)blockIdx.x * blockDim.x + threadIdx.x;
    const long long stride = (long long)gridDim.x * blockDim.x;
    for (; i < n4; i += stride) {
        const float4 wv = ((const float4*)w)[i];
        int packed = 0;
        float q;
        q = fminf(1.f, fmaxf(-1.f, rintf(wv.x * s))); packed |= ((int)q & 0xFF);
        q = fminf(1.f, fmaxf(-1.f, rintf(wv.y * s))); packed |= ((int)q & 0xFF) << 8;
        q = fminf(1.f, fmaxf(-1.f, rintf(wv.z * s))); packed |= ((int)q & 0xFF) << 16;
        q = fminf(1.f, fmaxf(-1.f, rintf(wv.w * s))); packed |= ((int)q & 0xFF) << 24;
        ((int*)wq)[i] = packed;
    }
}


template <int PER>
__global__ __launch_bounds__(256)
void rmsnorm_quant_kernel(const float* __restrict__ x, const float* __restrict__ g,
                          signed char* __restrict__ xq, float* __restrict__ inv_s,
                          int D) {
    __shared__ float red[256];
    const int row = blockIdx.x;
    const int base = threadIdx.x * PER;
    const float* xr = x + (long long)row * D;

    float v[PER];
    float ss = 0.0f;
#pragma unroll
    for (int i = 0; i < PER; ++i) { v[i] = xr[base + i]; ss += v[i] * v[i]; }
    red[threadIdx.x] = ss;
    __syncthreads();
    for (int o = 128; o > 0; o >>= 1) {
        if ((int)threadIdx.x < o) red[threadIdx.x] += red[threadIdx.x + o];
        __syncthreads();
    }
    const float rinv = rsqrtf(red[0] / (float)D + 1e-5f);
    __syncthreads();

    float amax = 0.0f;
#pragma unroll
    for (int i = 0; i < PER; ++i) {
        v[i] = v[i] * rinv * g[base + i];
        amax = fmaxf(amax, fabsf(v[i]));
    }
    red[threadIdx.x] = amax;
    __syncthreads();
    for (int o = 128; o > 0; o >>= 1) {
        if ((int)threadIdx.x < o) red[threadIdx.x] = fmaxf(red[threadIdx.x], red[threadIdx.x + o]);
        __syncthreads();
    }
    const float am = fmaxf(red[0], 1e-5f);
    const float s = 127.0f / am;
    if (threadIdx.x == 0) inv_s[row] = am / 127.0f;

    signed char* q = xq + (long long)row * D + base;
#pragma unroll
    for (int i = 0; i < PER; i += 4) {
        int packed = 0;
#pragma unroll
        for (int j = 0; j < 4; ++j) {
            float qf = fminf(127.0f, fmaxf(-128.0f, rintf(v[i + j] * s)));
            packed |= ((int)qf & 0xFF) << (8 * j);
        }
        *(int*)(q + i) = packed;
    }
}


template <bool FUSED>
__global__ __launch_bounds__(256)
void gemm_iu8_kernel(const signed char* __restrict__ A,
                     const signed char* __restrict__ Bg,
                     const signed char* __restrict__ Bv,
                     float* __restrict__ out,
                     const float* __restrict__ inv_sa,
                     const float* __restrict__ inv_sw,
                     int M, int N, int K) {
    constexpr int BM = 128, BN = 64, BK = 64;
    __shared__ signed char As[2][BM * BK];
    __shared__ signed char Bs[2][BN * BK];
    __shared__ signed char Bs2[2][BN * BK];

    const int tid  = threadIdx.x;
    const int lane = tid & 31;
    const int wave = tid >> 5;
    const int mw = wave & 3;
    const int nw = wave >> 2;
    const int hi  = lane >> 4;
    const int l16 = lane & 15;

    const int m0 = blockIdx.y * BM;
    const int n0 = blockIdx.x * BN;

    v8i accg[2][2], accv[2][2];
#pragma unroll
    for (int i = 0; i < 2; ++i)
#pragma unroll
        for (int j = 0; j < 2; ++j) { accg[i][j] = zero_v8i(); accv[i][j] = zero_v8i(); }

    const int arow = tid >> 1, ahalf = (tid & 1) * 32;
    const int brow = tid >> 2, bquart = (tid & 3) * 16;

    const int vaA = (m0 + arow) * K + ahalf;
    const int vaB = (n0 + brow) * K + bquart;

    const unsigned ldsA0 = (unsigned)(size_t)&As[0][arow * BK + ahalf];
    const unsigned ldsA1 = (unsigned)(size_t)&As[1][arow * BK + ahalf];
    const unsigned ldsB0 = (unsigned)(size_t)&Bs[0][brow * BK + bquart];
    const unsigned ldsB1 = (unsigned)(size_t)&Bs[1][brow * BK + bquart];
    const unsigned ldsC0 = (unsigned)(size_t)&Bs2[0][brow * BK + bquart];
    const unsigned ldsC1 = (unsigned)(size_t)&Bs2[1][brow * BK + bquart];

    auto issue_stage = [&](int buf, int kbase) {
        const unsigned la = buf ? ldsA1 : ldsA0;
        const unsigned lb = buf ? ldsB1 : ldsB0;
        async_b128(la, vaA, A + kbase);
        async_b128_off16(la, vaA, A + kbase);
        async_b128(lb, vaB, Bg + kbase);
        if (FUSED) {
            const unsigned lc = buf ? ldsC1 : ldsC0;
            async_b128(lc, vaB, Bv + kbase);
        }
    };

    auto compute_step = [&](int cur) {
        const signed char* Ac = &As[cur][0];
        const signed char* Bc = &Bs[cur][0];
        const signed char* Cc = &Bs2[cur][0];

        v8i afrag[2], bgfrag[2], bvfrag[2];
#pragma unroll
        for (int tm = 0; tm < 2; ++tm) {
            const signed char* p = Ac + (mw * 32 + tm * 16 + l16) * BK + hi * 16;
            const uint2 d0 = *(const uint2*)(p);
            const uint2 d1 = *(const uint2*)(p + 8);
            const uint2 d2 = *(const uint2*)(p + 32);
            const uint2 d3 = *(const uint2*)(p + 40);
            v8i a;
            a[0] = (int)d0.x; a[1] = (int)d0.y; a[2] = (int)d1.x; a[3] = (int)d1.y;
            a[4] = (int)d2.x; a[5] = (int)d2.y; a[6] = (int)d3.x; a[7] = (int)d3.y;
            afrag[tm] = a;
        }
#pragma unroll
        for (int tn = 0; tn < 2; ++tn) {
            const signed char* p = Bc + (nw * 32 + tn * 16 + l16) * BK + hi * 16;
            const uint4 d0 = *(const uint4*)(p);
            const uint4 d1 = *(const uint4*)(p + 32);
            v8i b;
            b[0] = (int)d0.x; b[1] = (int)d0.y; b[2] = (int)d0.z; b[3] = (int)d0.w;
            b[4] = (int)d1.x; b[5] = (int)d1.y; b[6] = (int)d1.z; b[7] = (int)d1.w;
            bgfrag[tn] = b;
            if (FUSED) {
                const signed char* p2 = Cc + (nw * 32 + tn * 16 + l16) * BK + hi * 16;
                const uint4 e0 = *(const uint4*)(p2);
                const uint4 e1 = *(const uint4*)(p2 + 32);
                v8i c;
                c[0] = (int)e0.x; c[1] = (int)e0.y; c[2] = (int)e0.z; c[3] = (int)e0.w;
                c[4] = (int)e1.x; c[5] = (int)e1.y; c[6] = (int)e1.z; c[7] = (int)e1.w;
                bvfrag[tn] = c;
            }
        }

#pragma unroll
        for (int tm = 0; tm < 2; ++tm)
#pragma unroll
            for (int tn = 0; tn < 2; ++tn) {
                accg[tm][tn] = __builtin_amdgcn_wmma_i32_16x16x64_iu8(
                    true, afrag[tm], true, bgfrag[tn], accg[tm][tn], false, false);
                if (FUSED)
                    accv[tm][tn] = __builtin_amdgcn_wmma_i32_16x16x64_iu8(
                        true, afrag[tm], true, bvfrag[tn], accv[tm][tn], false, false);
            }
    };

    const int nsteps = K / BK;
    issue_stage(0, 0);

    for (int s = 0; s < nsteps - 1; ++s) {
        const int cur = s & 1;
        issue_stage(cur ^ 1, (s + 1) * BK);
        if (FUSED) asm volatile("s_wait_asynccnt 0x4" ::: "memory");
        else       asm volatile("s_wait_asynccnt 0x3" ::: "memory");
        __syncthreads();
        compute_step(cur);
        __syncthreads();
    }

    asm volatile("s_wait_asynccnt 0x0" ::: "memory");
    __syncthreads();
    compute_step((nsteps - 1) & 1);

    const float sw = inv_sw[0];
#pragma unroll
    for (int tm = 0; tm < 2; ++tm)
#pragma unroll
        for (int tn = 0; tn < 2; ++tn)
#pragma unroll
            for (int r = 0; r < 8; ++r) {
                const int m = m0 + mw * 32 + tm * 16 + hi * 8 + r;
                const int n = n0 + nw * 32 + tn * 16 + l16;
                const float sc = inv_sa[m] * sw;
                const float yg = (float)accg[tm][tn][r] * sc;
                float val;
                if (FUSED) {
                    const float yv = (float)accv[tm][tn][r] * sc;
                    val = yg / (1.0f + __expf(-yg)) * yv;
                } else {
                    val = yg;
                }
                out[(long long)m * N + n] = val;
            }
}


extern "C" void kernel_launch(void* const* d_in, const int* in_sizes, int n_in,
                              void* d_out, int out_size, void* d_ws, size_t ws_size,
                              hipStream_t stream) {
    (void)in_sizes; (void)n_in; (void)out_size; (void)ws_size;

    const float* x      = (const float*)d_in[0];
    const float* w_gate = (const float*)d_in[1];
    const float* g_gate = (const float*)d_in[2];
    const float* w_down = (const float*)d_in[3];
    const float* g_down = (const float*)d_in[4];
    float* out = (float*)d_out;

    const int H = HDIM, I = IDIM, T = TOKENS;
    const long long n_wg = (long long)2 * I * H;
    const long long n_wd = (long long)H * I;

    char* ws = (char*)d_ws;
    float* red = (float*)ws;
    size_t off = 256;
    signed char* wq_gate = (signed char*)(ws + off); off += (size_t)n_wg;
    signed char* wq_down = (signed char*)(ws + off); off += (size_t)n_wd;
    signed char* xq      = (signed char*)(ws + off); off += (size_t)T * H;
    signed char* hq      = (signed char*)(ws + off); off += (size_t)T * I;
    float* inv_sa1 = (float*)(ws + off); off += (size_t)T * 4;
    float* inv_sa2 = (float*)(ws + off); off += (size_t)T * 4;
    float* h       = (float*)(ws + off); off += (size_t)T * I * 4;

    init_red_kernel<<<1, 32, 0, stream>>>(red);
    abs_sum_kernel<<<1024, 256, 0, stream>>>(w_gate, n_wg, red + 0);
    abs_sum_kernel<<<512, 256, 0, stream>>>(w_down, n_wd, red + 1);
    finalize_scales_kernel<<<1, 32, 0, stream>>>(red, n_wg, n_wd);

    quant_w_kernel<<<2048, 256, 0, stream>>>(w_gate, wq_gate, n_wg / 4, red + 2);
    quant_w_kernel<<<1024, 256, 0, stream>>>(w_down, wq_down, n_wd / 4, red + 4);

    rmsnorm_quant_kernel<HDIM / 256><<<T, 256, 0, stream>>>(x, g_gate, xq, inv_sa1, H);

    gemm_iu8_kernel<true><<<dim3(I / 64, T / 128), 256, 0, stream>>>(
        xq, wq_gate, wq_gate + (long long)I * H, h, inv_sa1, red + 3, T, I, H);

    rmsnorm_quant_kernel<IDIM / 256><<<T, 256, 0, stream>>>(h, g_down, hq, inv_sa2, I);

    gemm_iu8_kernel<false><<<dim3(H / 64, T / 128), 256, 0, stream>>>(
        hq, wq_down, nullptr, out, inv_sa2, red + 5, T, H, I);
}
